// BailingMoELinearDecoderLayer_50311246905461
// MI455X (gfx1250) — compile-verified, matrix operand order checked by reading, not run on hardware
//
#include <hip/hip_runtime.h>
#include <cstdint>
#include <cstddef>

// Problem dims
#define T_    1024
#define H_    2048
#define NH_   16
#define NKV_  4
#define HD_   128
#define E_    32
#define TOPK_ 4
#define I_    1024
#define EPS_  1e-6f

// Use the Tensor Data Mover for contiguous A-tile loads in the GEMM
#define USE_TDM 1

typedef unsigned short u16;
typedef __bf16 bf16x16 __attribute__((ext_vector_type(16)));
typedef float  f32x8   __attribute__((ext_vector_type(8)));
typedef unsigned uintx4 __attribute__((ext_vector_type(4)));
typedef int      intx4  __attribute__((ext_vector_type(4)));
typedef int      intx8  __attribute__((ext_vector_type(8)));

struct Frag32 { uint4 lo, hi; };

__device__ __forceinline__ u16 f2bf(float x) {
  unsigned u = __float_as_uint(x);
  u += 0x7fffu + ((u >> 16) & 1u);        // round-to-nearest-even
  return (u16)(u >> 16);
}

__device__ __forceinline__ bf16x16 ldfrag(const u16* p0, const u16* p1) {
  Frag32 r;
  r.lo = *(const uint4*)p0;
  r.hi = *(const uint4*)p1;
  return __builtin_bit_cast(bf16x16, r);
}

__device__ __forceinline__ f32x8 wmma_bf16(bf16x16 a, bf16x16 b, f32x8 c) {
#if defined(__gfx1250__)
  // 8 args: (neg_a, A, neg_b, B, c_mod, C, reuse_a, reuse_b)
  return __builtin_amdgcn_wmma_f32_16x16x32_bf16(false, a, false, b, (short)0, c, false, false);
#else
  (void)a; (void)b;
  return c;
#endif
}

#if defined(__gfx1250__) && USE_TDM
// TDM: DMA a 2D tile of bf16 (rows x row_elems) from global into LDS with
// per-row padding matching our LDS layout (row data + 16B pad per row).
// D# field packing per CDNA5 ISA.
// pad_icode: pad_interval code (interval = 2^(code+1) DWORDs); pad = 4 DW (16B).
__device__ __forceinline__ void tdm_load_2d_bf16(
    unsigned lds_off, const void* gaddr, int rows, int row_elems,
    int lda_elems, int tensor_d0, int pad_icode)
{
  size_t ga = (size_t)gaddr;
  uintx4 g0;
  g0[0] = 1u;                                            // count=1, user desc
  g0[1] = lds_off;                                       // lds_addr (bytes)
  g0[2] = (unsigned)(ga & 0xffffffffu);                  // global_addr[31:0]
  g0[3] = (unsigned)((ga >> 32) & 0x01ffffffu)           // global_addr[56:32]
        | (2u << 30);                                    // type=2 ("image")
  intx8 g1;
  g1[0] = (1 << 16)                                      // data_size = 2B
        | (1 << 20)                                      // pad_enable
        | (pad_icode << 22)                              // pad_interval
        | (3 << 25);                                     // pad_amount = 4 DW (16B)
  g1[1] = (tensor_d0 & 0xffff) << 16;                    // tensor_dim0[15:0]
  g1[2] = ((tensor_d0 >> 16) & 0xffff)                   // tensor_dim0[31:16]
        | ((rows & 0xffff) << 16);                       // tensor_dim1[15:0]
  g1[3] = ((rows >> 16) & 0xffff)                        // tensor_dim1[31:16]
        | ((row_elems & 0xffff) << 16);                  // tile_dim0
  g1[4] = (rows & 0xffff);                               // tile_dim1 (tile_dim2=0)
  g1[5] = lda_elems;                                     // tensor_dim0_stride[31:0]
  g1[6] = 0;                                             // stride hi / dim1_stride lo
  g1[7] = 0;
  intx4 gz = {0, 0, 0, 0};
#if __clang_major__ >= 23
  intx8 gz8 = {0, 0, 0, 0, 0, 0, 0, 0};
  __builtin_amdgcn_tensor_load_to_lds(g0, g1, gz, gz, gz8, 0);
#else
  __builtin_amdgcn_tensor_load_to_lds(g0, g1, gz, gz, 0);
#endif
}
#endif

__device__ __forceinline__ void wait_tensor0() {
#if defined(__gfx1250__) && USE_TDM
  __builtin_amdgcn_s_wait_tensorcnt(0);
#endif
}

// ---------------------------------------------------------------------------
// Generic bf16-WMMA GEMM: C[M x N] = A(bf16, row-major, lda) @ B(f32, row-major, ldb==N)
// Block tile 64(M) x 128(N), K-step 64. 256 threads = 8 waves (2 M x 4 N),
// each wave computes a 32x32 tile => 8 WMMAs per barrier interval.
// mode 0: C = acc
// mode 1: C = acc + resid            (residual add, same indexing as C)
// mode 2: MoE up-proj: A rows gathered via exp_list (g>>2 = token), C rows = g (slot)
// mode 3: MoE down-proj: A rows = g (slot), C rows = g
// A tiles: modes 0/1 use TDM (contiguous rows); modes 2/3 use manual gather.
// LDS row stride 72 elems (144B): 128B data + 16B pad, conflict-free, 16B aligned.
// ---------------------------------------------------------------------------
__global__ __launch_bounds__(256) void k_gemm(
    const u16* __restrict__ A, const float* __restrict__ Bbase, size_t Bstride,
    float* __restrict__ C, const float* __restrict__ resid,
    const int* __restrict__ exp_list, const int* __restrict__ cnt,
    int N, int K, int lda, int ldc, int mode)
{
  __shared__ __align__(16) u16 Alds[64 * 72];   // 64 rows x 64 (+8 pad) bf16
  __shared__ __align__(16) u16 Blds[128 * 72];  // 128 cols(n) x 64 (k) (+8 pad) = B^T

  const int tid  = threadIdx.x;
  const int lane = tid & 31, wid = tid >> 5;
  const int lr = lane & 15, hi = lane >> 4, kh = hi * 8;
  const int wm = (wid >> 2) * 32, wn = (wid & 3) * 32;
  const int m0 = blockIdx.y * 64, n0 = blockIdx.x * 128;
  const int e  = blockIdx.z;
  const float* B = Bbase + (size_t)e * Bstride;

  int cntE = 0;
  if (mode >= 2) { cntE = cnt[e]; if (m0 >= cntE) return; }   // uniform early-out

  f32x8 z = {0.f,0.f,0.f,0.f,0.f,0.f,0.f,0.f};
  f32x8 acc[2][2];
  acc[0][0] = z; acc[0][1] = z; acc[1][0] = z; acc[1][1] = z;

  // A-tile mapping: 4 threads per row, 16 bf16 (2 x 16B) each
  const int arowi = tid >> 2;
  const int acol  = (tid & 3) * 16;
  int arow = -1;
  if (mode >= 2) {
    int s = m0 + arowi;
    if (s < cntE) { int g = exp_list[e * T_ + s]; arow = (mode == 2) ? (g >> 2) : g; }
  } else {
    arow = m0 + arowi;
  }

  // B-tile mapping: thread covers 4 consecutive n x 8 k's (8 x b128 loads)
  const int bn4 = (tid & 31) * 4;       // 0..124
  const int bk8 = (tid >> 5) * 8;       // 0..56
  const unsigned a_lds_off = (unsigned)(size_t)(void*)&Alds[0];
  (void)a_lds_off;

  for (int k0 = 0; k0 < K; k0 += 64) {
    // --- A tile -> LDS ---
    if (mode <= 1) {
#if defined(__gfx1250__) && USE_TDM
      if (wid == 0)
        tdm_load_2d_bf16(a_lds_off, A + (size_t)m0 * lda + k0,
                         64, 64, lda, K - k0, /*pad_interval 32DW*/ 4);
#else
      const u16* src = A + (size_t)arow * lda + k0 + acol;
      *(uint4*)&Alds[arowi * 72 + acol]     = *(const uint4*)src;
      *(uint4*)&Alds[arowi * 72 + acol + 8] = *(const uint4*)(src + 8);
#endif
    } else {
      uint4 a0 = make_uint4(0u, 0u, 0u, 0u), a1 = a0;
      if (arow >= 0) {
        const u16* src = A + (size_t)arow * lda + k0 + acol;
        a0 = *(const uint4*)src;
        a1 = *(const uint4*)(src + 8);
      }
      *(uint4*)&Alds[arowi * 72 + acol]     = a0;
      *(uint4*)&Alds[arowi * 72 + acol + 8] = a1;
    }

    // --- B: global f32 (b128 loads) -> bf16, transposed into LDS as B^T[n][k] ---
    {
      float4 bv[8];
      #pragma unroll
      for (int kk = 0; kk < 8; ++kk)
        bv[kk] = *(const float4*)&B[(size_t)(k0 + bk8 + kk) * N + n0 + bn4];
      #pragma unroll
      for (int nn = 0; nn < 4; ++nn) {
        #pragma unroll
        for (int pp = 0; pp < 4; ++pp) {
          unsigned pk = (unsigned)f2bf(((const float*)&bv[2 * pp])[nn])
                      | ((unsigned)f2bf(((const float*)&bv[2 * pp + 1])[nn]) << 16);
          *(unsigned*)&Blds[(bn4 + nn) * 72 + bk8 + 2 * pp] = pk;
        }
      }
    }
    if (k0 + 64 < K)
      __builtin_prefetch(&B[(size_t)(k0 + 64 + bk8) * N + n0 + bn4], 0, 0);

    if (mode <= 1) wait_tensor0();      // TDM done before the barrier publishes
    __syncthreads();

    // --- fragments + WMMA: two 32-deep k-slices => 8 WMMAs per wave ---
    #pragma unroll
    for (int kc = 0; kc < 2; ++kc) {
      bf16x16 af[2], bfr[2];
      #pragma unroll
      for (int r = 0; r < 2; ++r) {
        const u16* p = &Alds[(wm + r * 16 + lr) * 72 + kc * 32 + kh];
        af[r] = ldfrag(p, p + 16);
      }
      #pragma unroll
      for (int c = 0; c < 2; ++c) {
        const u16* p = &Blds[(wn + c * 16 + lr) * 72 + kc * 32 + kh];
        bfr[c] = ldfrag(p, p + 16);
      }
      #pragma unroll
      for (int r = 0; r < 2; ++r)
        #pragma unroll
        for (int c = 0; c < 2; ++c)
          acc[r][c] = wmma_bf16(af[r], bfr[c], acc[r][c]);
    }
    __syncthreads();
  }

  // --- epilogue: C layout lane=(col), vgpr j=(row within 8-row half) ---
  #pragma unroll
  for (int r = 0; r < 2; ++r) {
    #pragma unroll
    for (int c = 0; c < 2; ++c) {
      int n = n0 + wn + c * 16 + lr;
      #pragma unroll
      for (int j = 0; j < 8; ++j) {
        int i = wm + r * 16 + hi * 8 + j;
        float v = acc[r][c][j];
        if (mode == 0) {
          C[(size_t)(m0 + i) * ldc + n] = v;
        } else if (mode == 1) {
          size_t idx = (size_t)(m0 + i) * ldc + n;
          C[idx] = v + resid[idx];
        } else {
          int s = m0 + i;
          if (s < cntE) {
            int g = exp_list[e * T_ + s];
            C[(size_t)g * ldc + n] = v;
          }
        }
      }
    }
  }
}

// ---------------------------------------------------------------------------
// RMSNorm: one block per token; writes bf16 (and optional f32 copy)
// ---------------------------------------------------------------------------
__global__ __launch_bounds__(256) void k_rmsnorm(
    const float* __restrict__ x, const float* __restrict__ w,
    u16* __restrict__ obf, float* __restrict__ of)
{
  __shared__ float red[8];
  const int t = blockIdx.x, tid = threadIdx.x;
  const float* xr = x + (size_t)t * H_;
  float v[8];
  float s = 0.f;
  #pragma unroll
  for (int i = 0; i < 8; ++i) { v[i] = xr[tid + 256 * i]; s += v[i] * v[i]; }
  #pragma unroll
  for (int off = 16; off; off >>= 1) s += __shfl_xor(s, off);
  if ((tid & 31) == 0) red[tid >> 5] = s;
  __syncthreads();
  float tot = 0.f;
  #pragma unroll
  for (int i = 0; i < 8; ++i) tot += red[i];
  const float sc = rsqrtf(tot / (float)H_ + EPS_);
  #pragma unroll
  for (int i = 0; i < 8; ++i) {
    int c = tid + 256 * i;
    float y = v[i] * sc * w[c];
    obf[(size_t)t * H_ + c] = f2bf(y);
    if (of) of[(size_t)t * H_ + c] = y;
  }
}

// ---------------------------------------------------------------------------
// RoPE + split qkv -> bf16 q/k/v.  qkv row: [q(2048) | k(512) | v(512)]
// ---------------------------------------------------------------------------
__global__ __launch_bounds__(256) void k_rope(
    const float* __restrict__ qkv, const int* __restrict__ pos,
    u16* __restrict__ qbf, u16* __restrict__ kbf, u16* __restrict__ vbf)
{
  const int t = blockIdx.x, tid = threadIdx.x;
  const float* row = qkv + (size_t)t * 3072;
  const float fp = (float)pos[t];
  const float c0 = 13.3046845f / 64.f;   // ln(600000)/half
  #pragma unroll
  for (int it = 0; it < 4; ++it) {       // q: 16 heads * 64 pairs
    int p = tid + 256 * it, h = p >> 6, i = p & 63;
    float inv = __expf(-(float)i * c0);
    float sn, cs; __sincosf(fp * inv, &sn, &cs);
    float x1 = row[h * HD_ + i], x2 = row[h * HD_ + 64 + i];
    qbf[((size_t)t * NH_ + h) * HD_ + i]      = f2bf(x1 * cs - x2 * sn);
    qbf[((size_t)t * NH_ + h) * HD_ + 64 + i] = f2bf(x2 * cs + x1 * sn);
  }
  {                                       // k: 4 heads * 64 pairs
    int h = tid >> 6, i = tid & 63;
    float inv = __expf(-(float)i * c0);
    float sn, cs; __sincosf(fp * inv, &sn, &cs);
    float x1 = row[NH_ * HD_ + h * HD_ + i], x2 = row[NH_ * HD_ + h * HD_ + 64 + i];
    kbf[((size_t)t * NKV_ + h) * HD_ + i]      = f2bf(x1 * cs - x2 * sn);
    kbf[((size_t)t * NKV_ + h) * HD_ + 64 + i] = f2bf(x2 * cs + x1 * sn);
  }
  #pragma unroll
  for (int it = 0; it < 2; ++it) {        // v: 4 heads * 128
    int p2 = tid + 256 * it, h = p2 >> 7, d = p2 & 127;
    vbf[((size_t)t * NKV_ + h) * HD_ + d] = f2bf(row[(NH_ + NKV_) * HD_ + h * HD_ + d]);
  }
}

// ---------------------------------------------------------------------------
// Flash attention: block = (64 q rows, 1 head), 4 waves; each wave owns 16 q rows.
// S = Q K^T and O += P V via WMMA bf16; online softmax in f32 VALU.
// ---------------------------------------------------------------------------
__global__ __launch_bounds__(128) void k_attn(
    const u16* __restrict__ qbf, const u16* __restrict__ kbf,
    const u16* __restrict__ vbf, u16* __restrict__ obf)
{
  __shared__ __align__(16) u16 Klds[32 * 136];      // [key][d], stride 136 (16B aligned)
  __shared__ __align__(16) u16 Vlds[128 * 40];      // [d][key] (transposed), stride 40
  __shared__ __align__(16) u16 Plds[4 * 16 * 40];   // per-wave P tile [row][key]

  const int tid = threadIdx.x, lane = tid & 31, wid = tid >> 5;
  const int lr = lane & 15, hi = lane >> 4, kh = hi * 8;
  const int head = blockIdx.y, kvh = head >> 2;     // NH/NKV == 4
  const int qbase = blockIdx.x * 64;
  const int qb = qbase + wid * 16;
  const float SCALE = 0.08838834764831845f;         // 1/sqrt(128)

  bf16x16 qf[4];
  {
    const u16* qp = qbf + ((size_t)(qb + lr) * NH_ + head) * HD_;
    #pragma unroll
    for (int c = 0; c < 4; ++c) qf[c] = ldfrag(qp + c * 32 + kh, qp + c * 32 + kh + 16);
  }

  f32x8 z = {0.f,0.f,0.f,0.f,0.f,0.f,0.f,0.f};
  f32x8 oacc[8];
  #pragma unroll
  for (int c = 0; c < 8; ++c) oacc[c] = z;
  float m_r[8], l_r[8];
  #pragma unroll
  for (int j = 0; j < 8; ++j) { m_r[j] = -1e30f; l_r[j] = 0.f; }

  u16* Pw = &Plds[wid * 16 * 40];
  const int nsteps = (blockIdx.x + 1) * 2;          // causal bound for this block

  for (int st = 0; st < nsteps; ++st) {
    const int key0 = st * 32;
    const int key = tid >> 2, db = (tid & 3) * 32;
    {   // K tile: 32 keys x 128 d, row-major (b128 loads/stores)
      const u16* src = kbf + ((size_t)(key0 + key) * NKV_ + kvh) * HD_ + db;
      u16* dst = &Klds[key * 136 + db];
      #pragma unroll
      for (int q4 = 0; q4 < 4; ++q4)
        *(uint4*)(dst + q4 * 8) = *(const uint4*)(src + q4 * 8);
    }
    {   // V tile transposed: Vlds[d][key]  (b128 loads, b16 LDS scatter)
      const u16* src = vbf + ((size_t)(key0 + key) * NKV_ + kvh) * HD_ + db;
      #pragma unroll
      for (int q4 = 0; q4 < 4; ++q4) {
        uint4 v = *(const uint4*)(src + q4 * 8);
        const u16* pv = (const u16*)&v;
        #pragma unroll
        for (int i = 0; i < 8; ++i)
          Vlds[(db + q4 * 8 + i) * 40 + key] = pv[i];
      }
    }
    __syncthreads();

    // S = Q K^T  (two 16x16 key subtiles, contraction over d=128 in 4 steps)
    f32x8 s0 = z, s1 = z;
    #pragma unroll
    for (int c = 0; c < 4; ++c) {
      const u16* p0 = &Klds[lr * 136 + c * 32 + kh];
      const u16* p1 = &Klds[(16 + lr) * 136 + c * 32 + kh];
      bf16x16 b0 = ldfrag(p0, p0 + 16);
      bf16x16 b1 = ldfrag(p1, p1 + 16);
      s0 = wmma_bf16(qf[c], b0, s0);
      s1 = wmma_bf16(qf[c], b1, s1);
    }

    // online softmax (row j lives in half `hi`, reductions across 16 lanes)
    #pragma unroll
    for (int j = 0; j < 8; ++j) {
      int qrow = qb + hi * 8 + j;
      float v0 = s0[j] * SCALE; if (key0 + lr > qrow)      v0 = -1e30f;
      float v1 = s1[j] * SCALE; if (key0 + 16 + lr > qrow) v1 = -1e30f;
      float mx = fmaxf(v0, v1);
      #pragma unroll
      for (int off = 8; off; off >>= 1) mx = fmaxf(mx, __shfl_xor(mx, off));
      float mn = fmaxf(m_r[j], mx);
      float p0 = __expf(v0 - mn), p1 = __expf(v1 - mn);
      float rs = p0 + p1;
      #pragma unroll
      for (int off = 8; off; off >>= 1) rs += __shfl_xor(rs, off);
      float sc = __expf(m_r[j] - mn);
      l_r[j] = l_r[j] * sc + rs;
      m_r[j] = mn;
      #pragma unroll
      for (int c = 0; c < 8; ++c) oacc[c][j] *= sc;
      Pw[(hi * 8 + j) * 40 + lr]      = f2bf(p0);
      Pw[(hi * 8 + j) * 40 + 16 + lr] = f2bf(p1);
    }

    // O += P V (contraction over 32 keys; 8 d-subtiles)
    {
      const u16* pp = &Pw[lr * 40 + kh];            // same-wave LDS ops are in-order
      bf16x16 pf = ldfrag(pp, pp + 16);
      #pragma unroll
      for (int c = 0; c < 8; ++c) {
        const u16* vp = &Vlds[(c * 16 + lr) * 40 + kh];
        bf16x16 vfr = ldfrag(vp, vp + 16);
        oacc[c] = wmma_bf16(pf, vfr, oacc[c]);
      }
    }
    __syncthreads();
  }

  #pragma unroll
  for (int c = 0; c < 8; ++c)
    #pragma unroll
    for (int j = 0; j < 8; ++j) {
      int t = qb + hi * 8 + j;
      float val = oacc[c][j] / l_r[j];
      obf[(size_t)t * H_ + head * HD_ + c * 16 + lr] = f2bf(val);
    }
}

// ---------------------------------------------------------------------------
// silu(g)*u -> bf16   (used for shared expert and MoE activations)
// ---------------------------------------------------------------------------
__global__ __launch_bounds__(256) void k_silu_mul(
    const float* __restrict__ g, const float* __restrict__ u,
    int ldg, int ldu, u16* __restrict__ out, int ncols)
{
  const int rowi = blockIdx.x, tid = threadIdx.x;
  const float* gr = g + (size_t)rowi * ldg;
  const float* ur = u + (size_t)rowi * ldu;
  u16* orow = out + (size_t)rowi * ncols;
  for (int c = tid; c < ncols; c += 256) {
    float gv = gr[c], uv = ur[c];
    float sg = gv / (1.f + __expf(-gv));
    orow[c] = f2bf(sg * uv);
  }
}

// ---------------------------------------------------------------------------
// Router: logits = h2 @ Wr^T, top-4, softmax over selected logits.
// Slot id g = t*4 + j (deterministic); per-expert compaction via atomic int only.
// ---------------------------------------------------------------------------
__global__ __launch_bounds__(256) void k_router(
    const float* __restrict__ h2f, const float* __restrict__ Wr,
    int* __restrict__ cnt, int* __restrict__ exp_list, float* __restrict__ tok_w)
{
  const int tid = threadIdx.x, lane = tid & 31, wid = tid >> 5;
  const int t = blockIdx.x * 8 + wid;
  const float* hr = h2f + (size_t)t * H_;

  float myl = 0.f;                       // lane e holds logit e
  for (int e = 0; e < E_; ++e) {
    const float* wr = Wr + (size_t)e * H_;
    float p = 0.f;
    for (int i = lane; i < H_; i += 32) p += hr[i] * wr[i];
    #pragma unroll
    for (int off = 16; off; off >>= 1) p += __shfl_xor(p, off);
    if (lane == e) myl = p;
  }

  bool taken = false;
  int te[4]; float tv[4];
  #pragma unroll
  for (int j = 0; j < 4; ++j) {          // argmax butterfly x4
    float v = taken ? -1e30f : myl;
    int idx = lane;
    #pragma unroll
    for (int off = 16; off; off >>= 1) {
      float ov = __shfl_xor(v, off);
      int   oi = __shfl_xor(idx, off);
      if (ov > v || (ov == v && oi < idx)) { v = ov; idx = oi; }
    }
    te[j] = idx; tv[j] = v;
    if (lane == idx) taken = true;
  }
  if (lane == 0) {
    float mx = tv[0], s = 0.f, w[4];
    #pragma unroll
    for (int j = 0; j < 4; ++j) { w[j] = __expf(tv[j] - mx); s += w[j]; }
    #pragma unroll
    for (int j = 0; j < 4; ++j) {
      int e  = te[j];
      int sl = atomicAdd(&cnt[e], 1);
      exp_list[e * T_ + sl] = t * 4 + j;
      tok_w[t * 4 + j] = w[j] / s;
    }
  }
}

__global__ void k_init(int* cnt) { if (threadIdx.x < E_) cnt[threadIdx.x] = 0; }

// out = x1 + shared + sum_j w_j * y[t*4+j]
__global__ __launch_bounds__(256) void k_final(
    const float* __restrict__ x1, const float* __restrict__ sharedf,
    const float* __restrict__ ybuf, const float* __restrict__ tok_w,
    float* __restrict__ out)
{
  const int t = blockIdx.x, tid = threadIdx.x;
  const float w0 = tok_w[t * 4 + 0], w1 = tok_w[t * 4 + 1];
  const float w2 = tok_w[t * 4 + 2], w3 = tok_w[t * 4 + 3];
  const float* y0 = ybuf + (size_t)(t * 4 + 0) * H_;
  const float* y1 = ybuf + (size_t)(t * 4 + 1) * H_;
  const float* y2 = ybuf + (size_t)(t * 4 + 2) * H_;
  const float* y3 = ybuf + (size_t)(t * 4 + 3) * H_;
  for (int c = tid; c < H_; c += 256) {
    float a = x1[(size_t)t * H_ + c] + sharedf[(size_t)t * H_ + c];
    a += w0 * y0[c] + w1 * y1[c] + w2 * y2[c] + w3 * y3[c];
    out[(size_t)t * H_ + c] = a;
  }
}

// ---------------------------------------------------------------------------
extern "C" void kernel_launch(void* const* d_in, const int* in_sizes, int n_in,
                              void* d_out, int out_size, void* d_ws, size_t ws_size,
                              hipStream_t stream)
{
  (void)in_sizes; (void)n_in; (void)out_size; (void)ws_size;
  const float* x     = (const float*)d_in[0];
  const int*   pos   = (const int*)  d_in[1];
  const float* Wqkv  = (const float*)d_in[2];
  const float* Wo    = (const float*)d_in[3];
  const float* ln1   = (const float*)d_in[4];
  const float* ln2   = (const float*)d_in[5];
  const float* Wr    = (const float*)d_in[6];
  const float* Wg    = (const float*)d_in[7];
  const float* Wu    = (const float*)d_in[8];
  const float* Wd    = (const float*)d_in[9];
  const float* Wshgu = (const float*)d_in[10];
  const float* Wshd  = (const float*)d_in[11];
  float* out = (float*)d_out;

  char* p = (char*)d_ws;
  auto alloc = [&](size_t bytes) -> char* {
    char* r = p; p += (bytes + 255) & ~(size_t)255; return r;
  };
  u16*   hbf     = (u16*)  alloc((size_t)T_ * H_ * 2);
  float* qkvf    = (float*)alloc((size_t)T_ * 3072 * 4);
  u16*   qbf     = (u16*)  alloc((size_t)T_ * NH_ * HD_ * 2);
  u16*   kbf     = (u16*)  alloc((size_t)T_ * NKV_ * HD_ * 2);
  u16*   vbf     = (u16*)  alloc((size_t)T_ * NKV_ * HD_ * 2);
  u16*   obf     = (u16*)  alloc((size_t)T_ * H_ * 2);
  float* x1f     = (float*)alloc((size_t)T_ * H_ * 4);
  u16*   h2bf    = (u16*)  alloc((size_t)T_ * H_ * 2);
  float* h2f     = (float*)alloc((size_t)T_ * H_ * 4);
  float* guf     = (float*)alloc((size_t)T_ * 2 * I_ * 4);
  u16*   sgubf   = (u16*)  alloc((size_t)T_ * I_ * 2);
  float* sharedf = (float*)alloc((size_t)T_ * H_ * 4);
  float* gbuf    = (float*)alloc((size_t)T_ * TOPK_ * I_ * 4);
  float* ubuf    = (float*)alloc((size_t)T_ * TOPK_ * I_ * 4);
  u16*   actbf   = (u16*)  alloc((size_t)T_ * TOPK_ * I_ * 2);
  float* ybuf    = (float*)alloc((size_t)T_ * TOPK_ * H_ * 4);
  int*   cnt     = (int*)  alloc((size_t)E_ * 4);
  int*   exlist  = (int*)  alloc((size_t)E_ * T_ * 4);
  float* tok_w   = (float*)alloc((size_t)T_ * TOPK_ * 4);

  k_init<<<1, 32, 0, stream>>>(cnt);

  // 1) h = rmsnorm(x, ln1) -> bf16
  k_rmsnorm<<<T_, 256, 0, stream>>>(x, ln1, hbf, nullptr);
  // 2) qkv = h @ Wqkv   (1024 x 2048 x 3072)
  k_gemm<<<dim3(3072 / 128, T_ / 64, 1), 256, 0, stream>>>(
      hbf, Wqkv, 0, qkvf, nullptr, nullptr, nullptr, 3072, H_, H_, 3072, 0);
  // 3) RoPE + split to bf16 q/k/v
  k_rope<<<T_, 256, 0, stream>>>(qkvf, pos, qbf, kbf, vbf);
  // 4) causal GQA flash attention
  k_attn<<<dim3(T_ / 64, NH_), 128, 0, stream>>>(qbf, kbf, vbf, obf);
  // 5) x1 = x + o @ Wo
  k_gemm<<<dim3(H_ / 128, T_ / 64, 1), 256, 0, stream>>>(
      obf, Wo, 0, x1f, x, nullptr, nullptr, H_, H_, H_, H_, 1);
  // 6) h2 = rmsnorm(x1, ln2) -> bf16 + f32
  k_rmsnorm<<<T_, 256, 0, stream>>>(x1f, ln2, h2bf, h2f);
  // 7) shared expert: gu = h2 @ Wsh_gu, sgu = silu(g)*u, shared = sgu @ Wsh_d
  k_gemm<<<dim3((2 * I_) / 128, T_ / 64, 1), 256, 0, stream>>>(
      h2bf, Wshgu, 0, guf, nullptr, nullptr, nullptr, 2 * I_, H_, H_, 2 * I_, 0);
  k_silu_mul<<<T_, 256, 0, stream>>>(guf, guf + I_, 2 * I_, 2 * I_, sgubf, I_);
  k_gemm<<<dim3(H_ / 128, T_ / 64, 1), 256, 0, stream>>>(
      sgubf, Wshd, 0, sharedf, nullptr, nullptr, nullptr, H_, I_, I_, H_, 0);
  // 8) routing: top-4 experts + normalized weights, per-expert token lists
  k_router<<<T_ / 8, 256, 0, stream>>>(h2f, Wr, cnt, exlist, tok_w);
  // 9) MoE grouped GEMMs (gathered rows, early-exit past cnt[e])
  k_gemm<<<dim3(I_ / 128, T_ / 64, E_), 256, 0, stream>>>(
      h2bf, Wg, (size_t)H_ * I_, gbuf, nullptr, exlist, cnt, I_, H_, H_, I_, 2);
  k_gemm<<<dim3(I_ / 128, T_ / 64, E_), 256, 0, stream>>>(
      h2bf, Wu, (size_t)H_ * I_, ubuf, nullptr, exlist, cnt, I_, H_, H_, I_, 2);
  k_silu_mul<<<T_ * TOPK_, 256, 0, stream>>>(gbuf, ubuf, I_, I_, actbf, I_);
  k_gemm<<<dim3(H_ / 128, T_ / 64, E_), 256, 0, stream>>>(
      actbf, Wd, (size_t)I_ * H_, ybuf, nullptr, exlist, cnt, H_, I_, I_, H_, 3);
  // 10) out = x1 + shared + sum_j w_j * y_j
  k_final<<<T_, 256, 0, stream>>>(x1f, sharedf, ybuf, tok_w, out);
}
